// Encoder_75436805587012
// MI455X (gfx1250) — matrix operand order repaired and checked by reading, not compiled, not run on hardware
//
// MI455X (gfx1250) transformer encoder forward.
// All GEMMs via v_wmma_f32_16x16x32_bf16 (wave32, f32 accum), register-blocked
// 32x64 per wave (2x4 fragments, 8 accumulators) for 4x A / 2x B reuse out of
// the 192MB L2. LN/softmax/bias epilogues in fp32 VALU. Workspace ~152 MB.
#include <hip/hip_runtime.h>
#include <hip/hip_bf16.h>

typedef __bf16 bf16;
typedef bf16  bf16x8 __attribute__((ext_vector_type(8)));
typedef bf16  v16bf  __attribute__((ext_vector_type(16)));
typedef float v8f    __attribute__((ext_vector_type(8)));

#define LN_EPS 1e-5f

__device__ __forceinline__ bf16 f2bf(float f) {
  unsigned u = __builtin_bit_cast(unsigned, f);
  unsigned r = u + 0x7FFFu + ((u >> 16) & 1u);   // round-to-nearest-even
  unsigned short h = (unsigned short)(r >> 16);
  return __builtin_bit_cast(bf16, h);
}
__device__ __forceinline__ float bf2f(bf16 h) {
  unsigned short u = __builtin_bit_cast(unsigned short, h);
  unsigned x = ((unsigned)u) << 16;
  return __builtin_bit_cast(float, x);
}

// two simultaneous block-wide sums (sum, sum-of-squares), 256 threads
__device__ __forceinline__ float2 blockSums2(float s, float q) {
  __shared__ float rs[256], rq[256];
  int tid = threadIdx.x;
  __syncthreads();                 // protect reuse across calls
  rs[tid] = s; rq[tid] = q;
  __syncthreads();
  for (int st = 128; st > 0; st >>= 1) {
    if (tid < st) { rs[tid] += rs[tid + st]; rq[tid] += rq[tid + st]; }
    __syncthreads();
  }
  float2 r; r.x = rs[0]; r.y = rq[0];
  __syncthreads();
  return r;
}

// ---------------------------------------------------------------------------
// Generalized batched WMMA GEMM:
//   C[z][m,n] = act( scale * sum_k A[z][m,k] * BT[z][n,k] + bias[n] )
// Per-z base offsets: off = (z>>4)*Hi + (z&15)*Lo  (covers b*strideB + h*strideH)
// Register blocking: each wave computes a 32x64 tile (2 M-frags x 4 N-frags,
// 8 f32 accumulators). Block = 8 waves stacked in M -> 256x64 block tile.
// Requires M % 256 == 0, N % 64 == 0, K % 32 == 0 (true for this workload).
// ---------------------------------------------------------------------------
__global__ void __launch_bounds__(256) wmma_gemm(
    const bf16* __restrict__ A, const bf16* __restrict__ BT,
    const float* __restrict__ bias, float* __restrict__ Cf, bf16* __restrict__ Cb,
    int M, int N, int K, int lda, int ldbt, int ldc,
    int act, float scale,
    long long aHi, long long aLo, long long bHi, long long bLo,
    long long cHi, long long cLo)
{
  int z = blockIdx.z;
  A  += (long long)(z >> 4) * aHi + (long long)(z & 15) * aLo;
  BT += (long long)(z >> 4) * bHi + (long long)(z & 15) * bLo;
  long long coff = (long long)(z >> 4) * cHi + (long long)(z & 15) * cLo;

  int wave = threadIdx.x >> 5;
  int lane = threadIdx.x & 31;
  int half = lane >> 4;          // 0: lanes 0-15, 1: lanes 16-31
  int lr   = lane & 15;

  int rowBase = blockIdx.y * 256 + wave * 32;   // 32 rows per wave
  int colBase = blockIdx.x * 64;                // 64 cols per wave
  if (rowBase >= M) return;

  // A fragment rows for this lane; B^T fragment rows (columns of B)
  const bf16* arow[2];
  const bf16* brow[4];
  #pragma unroll
  for (int mi = 0; mi < 2; ++mi)
    arow[mi] = A + (size_t)(rowBase + mi * 16 + lr) * lda;
  #pragma unroll
  for (int ni = 0; ni < 4; ++ni)
    brow[ni] = BT + (size_t)(colBase + ni * 16 + lr) * ldbt;

  v8f acc[2][4] = {};
  #pragma unroll 1
  for (int k0 = 0; k0 < K; k0 += 32) {
    // 16-bit A 16x32 layout: lanes<16 hold K {0..7,16..23}; lanes>=16 {8..15,24..31}
    union { bf16x8 h[2]; v16bf v; } fa[2], fb[4];
    #pragma unroll
    for (int mi = 0; mi < 2; ++mi) {
      fa[mi].h[0] = *(const bf16x8*)(arow[mi] + k0 + half * 8);
      fa[mi].h[1] = *(const bf16x8*)(arow[mi] + k0 + 16 + half * 8);
    }
    // 16-bit B 32x16 layout: lanes<16 hold K {0..7,16..23}; lanes>=16 {8..15,24..31}
    #pragma unroll
    for (int ni = 0; ni < 4; ++ni) {
      fb[ni].h[0] = *(const bf16x8*)(brow[ni] + k0 + half * 8);
      fb[ni].h[1] = *(const bf16x8*)(brow[ni] + k0 + 16 + half * 8);
    }
    if (k0 + 32 < K) {
      __builtin_prefetch(arow[0] + k0 + 32, 0, 1);   // global_prefetch_b8
      __builtin_prefetch(arow[1] + k0 + 32, 0, 1);
      __builtin_prefetch(brow[0] + k0 + 32, 0, 1);
      __builtin_prefetch(brow[2] + k0 + 32, 0, 1);
    }
    #pragma unroll
    for (int mi = 0; mi < 2; ++mi)
      #pragma unroll
      for (int ni = 0; ni < 4; ++ni)
        acc[mi][ni] = __builtin_amdgcn_wmma_f32_16x16x32_bf16(
            false, fa[mi].v, false, fb[ni].v, (short)0, acc[mi][ni], false, false);
  }

  // C/D layout: VGPR r -> M = r (+8 for lanes>=16), N = lane&15
  #pragma unroll
  for (int ni = 0; ni < 4; ++ni) {
    int n = colBase + ni * 16 + lr;
    float bv = bias ? bias[n] : 0.0f;
    #pragma unroll
    for (int mi = 0; mi < 2; ++mi) {
      #pragma unroll
      for (int r = 0; r < 8; ++r) {
        int m = rowBase + mi * 16 + r + half * 8;
        float v = acc[mi][ni][r] * scale + bv;
        if (act == 1)      v = v > 0.0f ? v : 0.0f;         // ReLU
        else if (act == 2) v = v > 0.0f ? v : 0.2f * v;     // LeakyReLU(0.2)
        long long o = coff + (long long)m * ldc + n;
        if (Cf) Cf[o] = v;
        if (Cb) Cb[o] = f2bf(v);
      }
    }
  }
}

// ---------------------------------------------------------------------------
// LN over VFS=2048 of x -> bf16  (one block per token, 8 elems/thread)
// ---------------------------------------------------------------------------
__global__ void __launch_bounds__(256) ln_x_kernel(
    const float* __restrict__ x, const float* __restrict__ g,
    const float* __restrict__ b, bf16* __restrict__ out)
{
  int t = blockIdx.x, tid = threadIdx.x;
  const float* xr = x + (size_t)t * 2048;
  float v[8], s = 0.f, q = 0.f;
  #pragma unroll
  for (int i = 0; i < 8; ++i) { float a = xr[tid + i * 256]; v[i] = a; s += a; q += a * a; }
  float2 r = blockSums2(s, q);
  float mean = r.x * (1.0f / 2048.0f);
  float var  = r.y * (1.0f / 2048.0f) - mean * mean;
  float inv  = rsqrtf(var + LN_EPS);
  bf16* orow = out + (size_t)t * 2048;
  #pragma unroll
  for (int i = 0; i < 8; ++i) {
    int c = tid + i * 256;
    orow[c] = f2bf((v[i] - mean) * inv * g[c] + b[c]);
  }
}

// ---------------------------------------------------------------------------
// y -> LN(g2,b2) -> + positional encoding -> LN(g3,b3) -> h (f32 + bf16)
// ---------------------------------------------------------------------------
__global__ void __launch_bounds__(256) embed_finish(
    const float* __restrict__ y,
    const float* __restrict__ g2, const float* __restrict__ b2,
    const float* __restrict__ g3, const float* __restrict__ b3,
    float* __restrict__ hf, bf16* __restrict__ hb)
{
  int t = blockIdx.x, tid = threadIdx.x;
  int s = t & 1023;                       // sequence position (B=2, S=1024)
  const float* yr = y + (size_t)t * 1024;
  float v[4], su = 0.f, sq = 0.f;
  #pragma unroll
  for (int i = 0; i < 4; ++i) { float a = yr[tid + i * 256]; v[i] = a; su += a; sq += a * a; }
  float2 r = blockSums2(su, sq);
  float mean = r.x * (1.0f / 1024.0f);
  float inv  = rsqrtf(r.y * (1.0f / 1024.0f) - mean * mean + LN_EPS);
  su = 0.f; sq = 0.f;
  #pragma unroll
  for (int i = 0; i < 4; ++i) {
    int c = tid + i * 256;
    float a = (v[i] - mean) * inv * g2[c] + b2[c];
    // pos enc: rate = 10000^(-(2*(c/2))/1024); even c -> sin, odd -> cos
    float rate = __expf(-(float)(c & ~1) * (9.210340371976184f / 1024.0f));
    float ang  = (float)s * rate;
    a += (c & 1) ? __cosf(ang) : __sinf(ang);
    v[i] = a; su += a; sq += a * a;
  }
  r = blockSums2(su, sq);
  float mean2 = r.x * (1.0f / 1024.0f);
  float inv2  = rsqrtf(r.y * (1.0f / 1024.0f) - mean2 * mean2 + LN_EPS);
  #pragma unroll
  for (int i = 0; i < 4; ++i) {
    int c = tid + i * 256;
    float o = (v[i] - mean2) * inv2 * g3[c] + b3[c];
    hf[(size_t)t * 1024 + c] = o;
    hb[(size_t)t * 1024 + c] = f2bf(o);
  }
}

// ---------------------------------------------------------------------------
// h = LN(res + y; g,b)  (per-thread in-place-safe; writes f32 + bf16)
// ---------------------------------------------------------------------------
__global__ void __launch_bounds__(256) add_ln(
    const float* __restrict__ res, const float* __restrict__ y,
    const float* __restrict__ g, const float* __restrict__ b,
    float* __restrict__ hf, bf16* __restrict__ hb)
{
  int t = blockIdx.x, tid = threadIdx.x;
  const float* rr = res + (size_t)t * 1024;
  const float* yr = y   + (size_t)t * 1024;
  float v[4], s = 0.f, q = 0.f;
  #pragma unroll
  for (int i = 0; i < 4; ++i) {
    int c = tid + i * 256;
    float a = rr[c] + yr[c];
    v[i] = a; s += a; q += a * a;
  }
  float2 r = blockSums2(s, q);
  float mean = r.x * (1.0f / 1024.0f);
  float inv  = rsqrtf(r.y * (1.0f / 1024.0f) - mean * mean + LN_EPS);
  #pragma unroll
  for (int i = 0; i < 4; ++i) {
    int c = tid + i * 256;
    float o = (v[i] - mean) * inv * g[c] + b[c];
    hf[(size_t)t * 1024 + c] = o;
    hb[(size_t)t * 1024 + c] = f2bf(o);
  }
}

// ---------------------------------------------------------------------------
// In-place row softmax on bf16 score buffer [32,1024,1024]; mask [2,1024]
// ---------------------------------------------------------------------------
__global__ void __launch_bounds__(256) softmax_rows(
    bf16* __restrict__ sp, const float* __restrict__ mask)
{
  int r = blockIdx.x;                 // 0 .. 32*1024-1
  int b = r >> 14;                    // batch = r / (16*1024)
  int tid = threadIdx.x;
  bf16* row = sp + (size_t)r * 1024;
  const float* mrow = mask + (size_t)b * 1024;
  float v[4], lmax = -3.0e38f;
  #pragma unroll
  for (int i = 0; i < 4; ++i) {
    int c = tid + i * 256;
    float xv = bf2f(row[c]) + mrow[c] * -1e9f;
    v[i] = xv; lmax = fmaxf(lmax, xv);
  }
  __shared__ float red[256];
  red[tid] = lmax; __syncthreads();
  for (int st = 128; st > 0; st >>= 1) {
    if (tid < st) red[tid] = fmaxf(red[tid], red[tid + st]);
    __syncthreads();
  }
  float m = red[0]; __syncthreads();
  float lsum = 0.f;
  #pragma unroll
  for (int i = 0; i < 4; ++i) { v[i] = __expf(v[i] - m); lsum += v[i]; }
  red[tid] = lsum; __syncthreads();
  for (int st = 128; st > 0; st >>= 1) {
    if (tid < st) red[tid] += red[tid + st];
    __syncthreads();
  }
  float inv = 1.0f / red[0];
  #pragma unroll
  for (int i = 0; i < 4; ++i) row[tid + i * 256] = f2bf(v[i] * inv);
}

// ---------------------------------------------------------------------------
// f32 [K,N] -> bf16 [N,K]  (coalesced writes)
// ---------------------------------------------------------------------------
__global__ void __launch_bounds__(256) convert_transpose(
    const float* __restrict__ in, bf16* __restrict__ out, int K, int N)
{
  size_t idx = (size_t)blockIdx.x * 256 + threadIdx.x;
  size_t total = (size_t)K * N;
  if (idx >= total) return;
  size_t n = idx / (size_t)K, k = idx % (size_t)K;
  out[idx] = f2bf(in[k * (size_t)N + n]);
}

// v bf16 [b,s][h*64+d] -> vT bf16 [b,h,d][s]   (2*16*64*1024 elements)
__global__ void __launch_bounds__(256) transpose_v(
    const bf16* __restrict__ v, bf16* __restrict__ vT)
{
  size_t idx = (size_t)blockIdx.x * 256 + threadIdx.x;
  int s = (int)(idx & 1023);
  size_t row = idx >> 10;
  int d = (int)(row & 63);
  int h = (int)((row >> 6) & 15);
  int b = (int)(row >> 10);
  vT[idx] = v[((size_t)(b * 1024 + s)) * 1024 + (size_t)h * 64 + d];
}

// ---------------------------------------------------------------------------
extern "C" void kernel_launch(void* const* d_in, const int* in_sizes, int n_in,
                              void* d_out, int out_size, void* d_ws, size_t ws_size,
                              hipStream_t stream)
{
  (void)in_sizes; (void)n_in; (void)out_size; (void)ws_size;
  const float* x     = (const float*)d_in[0];
  const float* mask  = (const float*)d_in[2];
  const float* eln1g = (const float*)d_in[3];
  const float* eln1b = (const float*)d_in[4];
  const float* emb_w = (const float*)d_in[5];
  const float* emb_b = (const float*)d_in[6];
  const float* eln2g = (const float*)d_in[7];
  const float* eln2b = (const float*)d_in[8];
  const float* eln3g = (const float*)d_in[9];
  const float* eln3b = (const float*)d_in[10];
  const float* wq = (const float*)d_in[11]; const float* bq = (const float*)d_in[12];
  const float* wk = (const float*)d_in[13]; const float* bk = (const float*)d_in[14];
  const float* wv = (const float*)d_in[15]; const float* bv = (const float*)d_in[16];
  const float* wo = (const float*)d_in[17]; const float* bo = (const float*)d_in[18];
  const float* w1 = (const float*)d_in[19]; const float* b1 = (const float*)d_in[20];
  const float* w2 = (const float*)d_in[21]; const float* b2 = (const float*)d_in[22];
  const float* l1g = (const float*)d_in[23]; const float* l1b = (const float*)d_in[24];
  const float* l2g = (const float*)d_in[25]; const float* l2b = (const float*)d_in[26];

  // ---- workspace layout (~152 MB total) ----
  char* w = (char*)d_ws;
  size_t o = 0;
  auto alloc = [&](size_t bytes) { void* p = w + o; o += (bytes + 255) & ~(size_t)255; return p; };
  bf16* XLN = (bf16*)alloc(2048ull * 2048 * 2);   // LN(x) bf16
  bf16* WQT = (bf16*)alloc(1024ull * 1024 * 2);   // per-layer W^T bf16 (reused)
  bf16* WKT = (bf16*)alloc(1024ull * 1024 * 2);   // (WQT+WKT also hold emb_w^T)
  bf16* WVT = (bf16*)alloc(1024ull * 1024 * 2);
  bf16* WOT = (bf16*)alloc(1024ull * 1024 * 2);
  bf16* W1T = (bf16*)alloc(4096ull * 1024 * 2);
  bf16* W2T = (bf16*)alloc(1024ull * 4096 * 2);
  float* H  = (float*)alloc(2048ull * 1024 * 4);  // residual stream f32
  bf16* HB  = (bf16*)alloc(2048ull * 1024 * 2);   // residual stream bf16
  bf16* QB  = (bf16*)alloc(2048ull * 1024 * 2);
  bf16* KB  = (bf16*)alloc(2048ull * 1024 * 2);
  bf16* VB  = (bf16*)alloc(2048ull * 1024 * 2);
  bf16* VT  = (bf16*)alloc(2048ull * 1024 * 2);   // [b,h,d,s]
  bf16* SP  = (bf16*)alloc(32ull * 1024 * 1024 * 2); // scores->probs, in place
  bf16* OB  = (bf16*)alloc(2048ull * 1024 * 2);   // attention context bf16
  float* Y  = (float*)alloc(2048ull * 1024 * 4);  // GEMM f32 output
  bf16* FB  = (bf16*)alloc(2048ull * 4096 * 2);   // FFN intermediate bf16

  auto GEMM = [&](const bf16* A, const bf16* BT, const float* bias,
                  float* Cf, bf16* Cb, int M, int N, int K,
                  int lda, int ldbt, int ldc, int act, float scale,
                  long long aHi, long long aLo, long long bHi, long long bLo,
                  long long cHi, long long cLo, int batches) {
    dim3 grid(N / 64, M / 256, batches);
    wmma_gemm<<<grid, 256, 0, stream>>>(A, BT, bias, Cf, Cb, M, N, K,
                                        lda, ldbt, ldc, act, scale,
                                        aHi, aLo, bHi, bLo, cHi, cLo);
  };

  // ---- embedding ----
  ln_x_kernel<<<2048, 256, 0, stream>>>(x, eln1g, eln1b, XLN);
  // emb_w [2048,1024] -> [1024,2048] bf16 into WQT..WKT region (4 MB)
  convert_transpose<<<8192, 256, 0, stream>>>(emb_w, WQT, 2048, 1024);
  GEMM(XLN, WQT, emb_b, Y, nullptr, 2048, 1024, 2048, 2048, 2048, 1024,
       /*relu*/1, 1.0f, 0, 0, 0, 0, 0, 0, 1);
  embed_finish<<<2048, 256, 0, stream>>>(Y, eln2g, eln2b, eln3g, eln3b, H, HB);

  // ---- 6 encoder layers ----
  for (int i = 0; i < 6; ++i) {
    size_t iD  = (size_t)i * 1024 * 1024;
    size_t iB  = (size_t)i * 1024;
    size_t iW1 = (size_t)i * 1024 * 4096;
    size_t iB1 = (size_t)i * 4096;

    convert_transpose<<<4096, 256, 0, stream>>>(wq + iD, WQT, 1024, 1024);
    convert_transpose<<<4096, 256, 0, stream>>>(wk + iD, WKT, 1024, 1024);
    convert_transpose<<<4096, 256, 0, stream>>>(wv + iD, WVT, 1024, 1024);
    convert_transpose<<<4096, 256, 0, stream>>>(wo + iD, WOT, 1024, 1024);
    convert_transpose<<<16384, 256, 0, stream>>>(w1 + iW1, W1T, 1024, 4096);
    convert_transpose<<<16384, 256, 0, stream>>>(w2 + iW1, W2T, 4096, 1024);

    // Q, K, V projections -> bf16
    GEMM(HB, WQT, bq + iB, nullptr, QB, 2048, 1024, 1024, 1024, 1024, 1024,
         0, 1.0f, 0, 0, 0, 0, 0, 0, 1);
    GEMM(HB, WKT, bk + iB, nullptr, KB, 2048, 1024, 1024, 1024, 1024, 1024,
         0, 1.0f, 0, 0, 0, 0, 0, 0, 1);
    GEMM(HB, WVT, bv + iB, nullptr, VB, 2048, 1024, 1024, 1024, 1024, 1024,
         0, 1.0f, 0, 0, 0, 0, 0, 0, 1);
    transpose_v<<<8192, 256, 0, stream>>>(VB, VT);

    // scores = scale * Q Kᵀ   (z = b*16+h; per-z A/B offsets b*1M + h*64)
    GEMM(QB, KB, nullptr, nullptr, SP, 1024, 1024, 64, 1024, 1024, 1024,
         0, 0.125f, 1048576, 64, 1048576, 64, 16777216, 1048576, 32);
    softmax_rows<<<32768, 256, 0, stream>>>(SP, mask);
    // context = P V   (C scattered back head-interleaved: b*1M + h*64)
    GEMM(SP, VT, nullptr, nullptr, OB, 1024, 64, 1024, 1024, 1024, 1024,
         0, 1.0f, 16777216, 1048576, 1048576, 65536, 1048576, 64, 32);
    // output projection
    GEMM(OB, WOT, bo + iB, Y, nullptr, 2048, 1024, 1024, 1024, 1024, 1024,
         0, 1.0f, 0, 0, 0, 0, 0, 0, 1);
    add_ln<<<2048, 256, 0, stream>>>(H, Y, l1g + iB, l1b + iB, H, HB);

    // FFN: D -> 4D (LeakyReLU fused) -> D
    GEMM(HB, W1T, b1 + iB1, nullptr, FB, 2048, 4096, 1024, 1024, 1024, 4096,
         /*leaky*/2, 1.0f, 0, 0, 0, 0, 0, 0, 1);
    GEMM(FB, W2T, b2 + iB, Y, nullptr, 2048, 1024, 4096, 4096, 4096, 1024,
         0, 1.0f, 0, 0, 0, 0, 0, 0, 1);
    float* dst = (i == 5) ? (float*)d_out : H;
    add_ln<<<2048, 256, 0, stream>>>(H, Y, l2g + iB, l2b + iB, dst, HB);
  }
}
